// RNNSeq2Seq_63840393887986
// MI455X (gfx1250) — matrix operand order repaired and checked by reading, not compiled, not run on hardware
//
#include <hip/hip_runtime.h>
#include <hip/hip_bf16.h>
#include <math.h>

typedef __bf16 bf16;
typedef __attribute__((ext_vector_type(16))) __bf16 v16bf;
typedef __attribute__((ext_vector_type(8)))  float  v8f;

#define H      512
#define G3     1536
#define NPOS   64
#define T_IN   800
#define T_OUT  200
#define C_IN   80
#define NC     80
#define NFILT  32
#define KSZ    31
#define B      32
#define ROWS_ENC (T_IN*B)   /* 25600 */

union AFrag { v16bf v; bf16 h[16]; uint4 q[2]; };

__device__ __forceinline__ v8f wmma_bf16(v16bf a, v16bf b, v8f c) {
  return __builtin_amdgcn_wmma_f32_16x16x32_bf16(false, a, false, b, (short)0, c, false, false);
}
__device__ __forceinline__ float sigm(float x) { return 1.f / (1.f + __expf(-x)); }

// ---------------------------------------------------------------------------
// Pack weight W (N x K, row-major f32, computing X @ W^T) into WMMA B-fragment
// layout: tile (nt, ks) holds B[k0..k0+31][n0..n0+15]; within tile, element
// index = lane*16 + e with n = n0 + (lane&15), k = k0 + (lane>=16?8:0) + (e<8 ? e : 16+e-8).
// ---------------------------------------------------------------------------
__global__ void k_pack(const float* __restrict__ W, bf16* __restrict__ P,
                       int K, int N, int Kpad, int Npad) {
  int idx = blockIdx.x * blockDim.x + threadIdx.x;
  int ksteps = Kpad >> 5;
  int total = (Npad >> 4) * ksteps * 512;
  if (idx >= total) return;
  int tile = idx >> 9;
  int w    = idx & 511;
  int lane = w >> 4;
  int e    = w & 15;
  int nt   = tile / ksteps;
  int ks   = tile - nt * ksteps;
  int n    = nt * 16 + (lane & 15);
  int k    = ks * 32 + ((lane & 16) ? 8 : 0) + ((e < 8) ? e : (16 + e - 8));
  float v  = (k < K && n < N) ? W[(size_t)n * K + k] : 0.f;
  P[idx] = (bf16)v;
}

// ---------------------------------------------------------------------------
// Generic GEMM: C(rows x ntiles*16) = A(rows x Kpad, bf16 rowmajor) @ Wpacked
// (+bias, +optional ELU, +optional bf16 mirror). One wave per 16x16 tile.
// ---------------------------------------------------------------------------
__global__ void k_gemm(const bf16* __restrict__ A, int lda,
                       const bf16* __restrict__ Wp, const float* __restrict__ bias,
                       float* __restrict__ C, int ldc,
                       bf16* __restrict__ Cb, int ldcb,
                       int rows, int ksteps, int ntiles, int act) {
  int wid  = blockIdx.x * (blockDim.x >> 5) + (threadIdx.x >> 5);
  int lane = threadIdx.x & 31;
  int tiles = (rows >> 4) * ntiles;
  if (wid >= tiles) return;
  int mt = wid / ntiles;
  int nt = wid - mt * ntiles;
  const bf16* arow = A + (size_t)(mt * 16 + (lane & 15)) * lda;
  const bf16* bptr = Wp + (size_t)nt * ksteps * 512 + lane * 16;
  int koff = (lane & 16) ? 8 : 0;
  v8f acc = {0.f,0.f,0.f,0.f,0.f,0.f,0.f,0.f};
  for (int ks = 0; ks < ksteps; ++ks) {
    AFrag a;
    a.q[0] = *(const uint4*)(arow + ks * 32 + koff);
    a.q[1] = *(const uint4*)(arow + ks * 32 + 16 + koff);
    v16bf bfr = *(const v16bf*)(bptr + (size_t)ks * 512);
    acc = wmma_bf16(a.v, bfr, acc);
  }
  int n = nt * 16 + (lane & 15);
  int mbase = mt * 16 + ((lane & 16) ? 8 : 0);
  float bv = bias ? bias[n] : 0.f;
#pragma unroll
  for (int r = 0; r < 8; ++r) {
    float v = acc[r] + bv;
    if (act == 1) v = (v > 0.f) ? v : (__expf(v) - 1.f);  // ELU
    int row = mbase + r;
    C[(size_t)row * ldc + n] = v;
    if (Cb) Cb[(size_t)row * ldcb + n] = (bf16)v;
  }
}

// ---------------------------------------------------------------------------
// Build encoder input: xin[(t*B+b)][c] = flip(x)_c | pe_in(t) | zero pad to 160
// ---------------------------------------------------------------------------
__global__ void k_prep_xin(const float* __restrict__ x, bf16* __restrict__ xin) {
  int idx = blockIdx.x * blockDim.x + threadIdx.x;
  const int total = ROWS_ENC * 160;
  if (idx >= total) return;
  int c = idx % 160;
  int row = idx / 160;
  int b = row & 31;
  int t = row >> 5;
  float v;
  if (c < C_IN) {
    v = x[(size_t)b * T_IN * C_IN + (size_t)t * C_IN + (C_IN - 1 - c)];
  } else if (c < C_IN + NPOS) {
    int d = c - C_IN;
    int i = d >> 1;
    float div = __expf(-(float)(2 * i) * (logf(10000.f) / (float)NPOS));
    float arg = (float)t * div;
    v = (d & 1) ? __cosf(arg) : __sinf(arg);
  } else {
    v = 0.f;
  }
  xin[idx] = (bf16)v;
}

// ---------------------------------------------------------------------------
// Persistent GRU recurrence: grid.x = 2 (forward / backward direction).
// One 1024-thread workgroup (32 waves); h (32x512 f32) and gh (32x1536 f32)
// live in LDS (256 KB dynamic — needs CDNA5's 320 KB WGP LDS).
// Each wave owns 6 of the 192 output tiles, loops 16 k-steps of WMMA.
// ---------------------------------------------------------------------------
__global__ __launch_bounds__(1024) void k_gru_rec(
    const float* __restrict__ giF, const float* __restrict__ giB,
    const bf16* __restrict__ WpF, const bf16* __restrict__ WpB,
    const float* __restrict__ bhhF, const float* __restrict__ bhhB,
    bf16* __restrict__ yout, bf16* __restrict__ hflat,
    int flatOffF, int flatOffB) {
  extern __shared__ char smem[];
  float* hS  = (float*)smem;        // 32*512
  float* ghS = hS + B * H;          // 32*1536
  const int dir = blockIdx.x;
  const float* gi  = dir ? giB : giF;
  const bf16*  Wp  = dir ? WpB : WpF;
  const float* bhh = dir ? bhhB : bhhF;
  const int colOff  = dir ? H : 0;
  const int flatOff = dir ? flatOffB : flatOffF;
  const int tid  = threadIdx.x;
  const int lane = tid & 31;
  const int wid  = tid >> 5;
  const int mt   = wid >> 4;        // 0..1
  const int ngrp = wid & 15;        // 0..15 -> nt = ngrp*6 + j
  for (int i = tid; i < B * H; i += 1024) hS[i] = 0.f;
  __syncthreads();
  const int koff  = (lane & 16) ? 8 : 0;
  const int m     = mt * 16 + (lane & 15);
  const float* hrow = hS + m * H;
  const int mbase = mt * 16 + ((lane & 16) ? 8 : 0);
  const v8f z8 = {0.f,0.f,0.f,0.f,0.f,0.f,0.f,0.f};
  for (int s = 0; s < T_IN; ++s) {
    int t = dir ? (T_IN - 1 - s) : s;
    v8f acc[6];
#pragma unroll
    for (int j = 0; j < 6; ++j) acc[j] = z8;
    for (int ks = 0; ks < 16; ++ks) {
      AFrag a;
      const float* hp = hrow + ks * 32 + koff;
#pragma unroll
      for (int i = 0; i < 8; ++i) a.h[i] = (bf16)hp[i];
#pragma unroll
      for (int i = 0; i < 8; ++i) a.h[8 + i] = (bf16)hp[16 + i];
#pragma unroll
      for (int j = 0; j < 6; ++j) {
        int nt = ngrp * 6 + j;
        v16bf bfr = *(const v16bf*)(Wp + (size_t)(nt * 16 + ks) * 512 + lane * 16);
        acc[j] = wmma_bf16(a.v, bfr, acc[j]);
      }
    }
#pragma unroll
    for (int j = 0; j < 6; ++j) {
      int n = (ngrp * 6 + j) * 16 + (lane & 15);
#pragma unroll
      for (int r = 0; r < 8; ++r) ghS[(mbase + r) * G3 + n] = acc[j][r];
    }
    __syncthreads();
    const float* girow = gi + (size_t)t * B * G3;
    for (int i = tid; i < B * H; i += 1024) {
      int bb = i >> 9, u = i & 511;
      float ghr = ghS[bb * G3 + u]         + bhh[u];
      float ghz = ghS[bb * G3 + H + u]     + bhh[H + u];
      float ghn = ghS[bb * G3 + 2 * H + u] + bhh[2 * H + u];
      float gir = girow[bb * G3 + u];
      float giz = girow[bb * G3 + H + u];
      float gin = girow[bb * G3 + 2 * H + u];
      float r  = sigm(gir + ghr);
      float z  = sigm(giz + ghz);
      float nn = tanhf(gin + r * ghn);
      float hnew = (1.f - z) * nn + z * hS[i];
      hS[i] = hnew;
      yout[((size_t)t * B + bb) * (2 * H) + colOff + u] = (bf16)hnew;
    }
    __syncthreads();
  }
  for (int i = tid; i < B * H; i += 1024) {
    int bb = i >> 9, u = i & 511;
    hflat[bb * 2048 + flatOff + u] = (bf16)hS[i];
  }
}

// hdec_tmp (B x 1024) -> hd (2 x B x 512) f32 + bf16 mirror
__global__ void k_split_hdec(const float* __restrict__ tmp,
                             float* __restrict__ hd, bf16* __restrict__ hdb) {
  int idx = blockIdx.x * blockDim.x + threadIdx.x;
  if (idx >= B * 2 * H) return;
  int b = idx >> 10;
  int c = idx & 1023;
  int l = c >> 9;
  int u = c & 511;
  float v = tmp[idx];
  hd[((size_t)l * B + b) * H + u]  = v;
  hdb[((size_t)l * B + b) * H + u] = (bf16)v;
}

__global__ void k_zero(float* __restrict__ p, int n) {
  int i = blockIdx.x * blockDim.x + threadIdx.x;
  if (i < n) p[i] = 0.f;
}

// location conv: loc[(t*B+b)][f] = sum_k W[f][0][k]*a[t+k-15][b] + W[f][1][k]*ac[...]
__global__ void k_loc_conv(const float* __restrict__ a, const float* __restrict__ ac,
                           const float* __restrict__ Wc, bf16* __restrict__ loc) {
  int idx = blockIdx.x * blockDim.x + threadIdx.x;
  if (idx >= ROWS_ENC * NFILT) return;
  int f = idx & 31;
  int row = idx >> 5;
  int b = row & 31;
  int t = row >> 5;
  float s = 0.f;
#pragma unroll
  for (int k = 0; k < KSZ; ++k) {
    int tt = t + k - (KSZ - 1) / 2;
    if (tt >= 0 && tt < T_IN) {
      s += Wc[f * 2 * KSZ + k]       * a[tt * B + b];
      s += Wc[f * 2 * KSZ + KSZ + k] * ac[tt * B + b];
    }
  }
  loc[idx] = (bf16)s;
}

// Fused: locs = loc @ loc_dense^T (WMMA, K=32), e = tanh(q+kv+locs) . v
// One wave per 16-row tile of (t*B+b); reduce over lanes with shfl_xor.
__global__ void k_energy(const bf16* __restrict__ loc, const bf16* __restrict__ pld,
                         const float* __restrict__ q, const float* __restrict__ kv,
                         const float* __restrict__ vW, float* __restrict__ e,
                         int mtiles) {
  int wid  = blockIdx.x * (blockDim.x >> 5) + (threadIdx.x >> 5);
  int lane = threadIdx.x & 31;
  if (wid >= mtiles) return;
  int mrow = wid * 16 + (lane & 15);
  int koff = (lane & 16) ? 8 : 0;
  AFrag a;
  a.q[0] = *(const uint4*)(loc + (size_t)mrow * 32 + koff);
  a.q[1] = *(const uint4*)(loc + (size_t)mrow * 32 + 16 + koff);
  int mbase = wid * 16 + ((lane & 16) ? 8 : 0);
  float sums[8] = {0.f,0.f,0.f,0.f,0.f,0.f,0.f,0.f};
  const v8f z8 = {0.f,0.f,0.f,0.f,0.f,0.f,0.f,0.f};
  for (int nt = 0; nt < 32; ++nt) {
    v16bf bfr = *(const v16bf*)(pld + (size_t)nt * 512 + lane * 16);
    v8f acc = wmma_bf16(a.v, bfr, z8);
    int n = nt * 16 + (lane & 15);
    float vn = vW[n];
#pragma unroll
    for (int r = 0; r < 8; ++r) {
      int row = mbase + r;
      int b = row & 31;
      float tv = tanhf(acc[r] + q[b * H + n] + kv[(size_t)row * H + n]);
      sums[r] += vn * tv;
    }
  }
#pragma unroll
  for (int off = 8; off >= 1; off >>= 1) {
#pragma unroll
    for (int r = 0; r < 8; ++r) sums[r] += __shfl_xor(sums[r], off, 32);
  }
  if ((lane & 15) == 0) {
#pragma unroll
    for (int r = 0; r < 8; ++r) e[mbase + r] = sums[r];
  }
}

// softmax over t, a=scores, ac+=scores, ctx = sum_t scores*kv; write ctx into
// feat[:,512:1024] and decin[:,144:656] bf16.
__global__ void k_softmax_ctx(const float* __restrict__ e, float* __restrict__ a,
                              float* __restrict__ ac, const float* __restrict__ kv,
                              float* __restrict__ ctx, bf16* __restrict__ feat,
                              bf16* __restrict__ decin) {
  __shared__ float red[256];
  int b = blockIdx.x;
  int tid = threadIdx.x;
  float m = -1e30f;
  for (int t = tid; t < T_IN; t += 256) m = fmaxf(m, e[t * B + b]);
  red[tid] = m; __syncthreads();
  for (int s = 128; s > 0; s >>= 1) { if (tid < s) red[tid] = fmaxf(red[tid], red[tid + s]); __syncthreads(); }
  m = red[0]; __syncthreads();
  float sum = 0.f;
  for (int t = tid; t < T_IN; t += 256) sum += __expf(e[t * B + b] - m);
  red[tid] = sum; __syncthreads();
  for (int s = 128; s > 0; s >>= 1) { if (tid < s) red[tid] += red[tid + s]; __syncthreads(); }
  float inv = 1.f / red[0]; __syncthreads();
  for (int t = tid; t < T_IN; t += 256) {
    float sc = __expf(e[t * B + b] - m) * inv;
    a[t * B + b] = sc;
    ac[t * B + b] += sc;
  }
  __syncthreads();
  for (int h = tid; h < H; h += 256) {
    float s2 = 0.f;
    for (int t = 0; t < T_IN; ++t) s2 += a[t * B + b] * kv[((size_t)t * B + b) * H + h];
    ctx[b * H + h] = s2;
    feat[b * 1024 + H + h]  = (bf16)s2;
    decin[b * 672 + 144 + h] = (bf16)s2;
  }
}

// decin cols 0..79 = prev logits (0 at t==0), 80..143 = pe_out[0], 656..671 = 0
__global__ void k_build_decin(bf16* __restrict__ decin, const float* __restrict__ out, int t) {
  int idx = blockIdx.x * blockDim.x + threadIdx.x;
  if (idx >= B * 672) return;
  int b = idx / 672;
  int c = idx - b * 672;
  if (c < NC) {
    float v = (t > 0) ? out[(size_t)b * T_OUT * NC + (size_t)(t - 1) * NC + c] : 0.f;
    decin[idx] = (bf16)v;
  } else if (c < NC + NPOS) {
    int d = c - NC;
    decin[idx] = (bf16)((d & 1) ? 1.f : 0.f);  // pe_out[0]: sin(0)=0, cos(0)=1
  } else if (c >= 656) {
    decin[idx] = (bf16)0.f;
  }
}

// GRU gate combine for decoder (biases already folded into gi/gh GEMMs)
__global__ void k_gate(const float* __restrict__ gi, const float* __restrict__ gh,
                       float* __restrict__ h, bf16* __restrict__ hb,
                       bf16* __restrict__ extra, int extraLd) {
  int idx = blockIdx.x * blockDim.x + threadIdx.x;
  if (idx >= B * H) return;
  int b = idx >> 9, u = idx & 511;
  float r  = sigm(gi[b * G3 + u] + gh[b * G3 + u]);
  float z  = sigm(gi[b * G3 + H + u] + gh[b * G3 + H + u]);
  float nn = tanhf(gi[b * G3 + 2 * H + u] + r * gh[b * G3 + 2 * H + u]);
  float hv = (1.f - z) * nn + z * h[idx];
  h[idx]  = hv;
  hb[idx] = (bf16)hv;
  if (extra) extra[(size_t)b * extraLd + u] = (bf16)hv;
}

// ---------------------------------------------------------------------------
extern "C" void kernel_launch(void* const* d_in, const int* in_sizes, int n_in,
                              void* d_out, int out_size, void* d_ws, size_t ws_size,
                              hipStream_t stream) {
  (void)in_sizes; (void)n_in; (void)out_size; (void)ws_size;
  // ---- inputs (setup_inputs dict order: x, then params insertion order) ----
  const float* x       = (const float*)d_in[0];
  const float* e0fWih  = (const float*)d_in[1];
  const float* e0fWhh  = (const float*)d_in[2];
  const float* e0fbih  = (const float*)d_in[3];
  const float* e0fbhh  = (const float*)d_in[4];
  const float* e0bWih  = (const float*)d_in[5];
  const float* e0bWhh  = (const float*)d_in[6];
  const float* e0bbih  = (const float*)d_in[7];
  const float* e0bbhh  = (const float*)d_in[8];
  const float* e1fWih  = (const float*)d_in[9];
  const float* e1fWhh  = (const float*)d_in[10];
  const float* e1fbih  = (const float*)d_in[11];
  const float* e1fbhh  = (const float*)d_in[12];
  const float* e1bWih  = (const float*)d_in[13];
  const float* e1bWhh  = (const float*)d_in[14];
  const float* e1bbih  = (const float*)d_in[15];
  const float* e1bbhh  = (const float*)d_in[16];
  const float* h2hW    = (const float*)d_in[17];
  const float* h2hb    = (const float*)d_in[18];
  const float* memW    = (const float*)d_in[19];
  const float* queryW  = (const float*)d_in[20];
  const float* vW      = (const float*)d_in[21];
  const float* locW    = (const float*)d_in[22];
  const float* locDW   = (const float*)d_in[23];
  const float* d0Wih   = (const float*)d_in[24];
  const float* d0Whh   = (const float*)d_in[25];
  const float* d0bih   = (const float*)d_in[26];
  const float* d0bhh   = (const float*)d_in[27];
  const float* d1Wih   = (const float*)d_in[28];
  const float* d1Whh   = (const float*)d_in[29];
  const float* d1bih   = (const float*)d_in[30];
  const float* d1bhh   = (const float*)d_in[31];
  const float* clsW1   = (const float*)d_in[32];
  const float* clsb1   = (const float*)d_in[33];
  const float* clsW2   = (const float*)d_in[34];
  const float* clsb2   = (const float*)d_in[35];
  float* out = (float*)d_out;

  // ---- workspace carve-out ----
  char* base = (char*)d_ws;
  size_t off = 0;
  auto alloc = [&](size_t bytes) -> char* {
    char* p = base + off;
    off = (off + bytes + 255) & ~(size_t)255;
    return p;
  };
  bf16*  xin    = (bf16*) alloc((size_t)ROWS_ENC * 160 * 2);
  float* giF    = (float*)alloc((size_t)ROWS_ENC * G3 * 4);
  float* giB    = (float*)alloc((size_t)ROWS_ENC * G3 * 4);
  bf16*  ycur   = (bf16*) alloc((size_t)ROWS_ENC * 1024 * 2);
  float* kv     = (float*)alloc((size_t)ROWS_ENC * H * 4);
  bf16*  hflat  = (bf16*) alloc((size_t)B * 2048 * 2);
  float* hdecT  = (float*)alloc((size_t)B * 1024 * 4);
  float* hd     = (float*)alloc((size_t)2 * B * H * 4);
  bf16*  hdb    = (bf16*) alloc((size_t)2 * B * H * 2);
  float* qbuf   = (float*)alloc((size_t)B * H * 4);
  float* aBuf   = (float*)alloc((size_t)ROWS_ENC * 4);
  float* acBuf  = (float*)alloc((size_t)ROWS_ENC * 4);
  float* eBuf   = (float*)alloc((size_t)ROWS_ENC * 4);
  bf16*  locBuf = (bf16*) alloc((size_t)ROWS_ENC * NFILT * 2);
  float* ctx    = (float*)alloc((size_t)B * H * 4);
  bf16*  decin  = (bf16*) alloc((size_t)B * 672 * 2);
  float* giD    = (float*)alloc((size_t)B * G3 * 4);
  float* ghD    = (float*)alloc((size_t)B * G3 * 4);
  bf16*  feat   = (bf16*) alloc((size_t)B * 1024 * 2);
  float* hid    = (float*)alloc((size_t)B * H * 4);
  bf16*  hidb   = (bf16*) alloc((size_t)B * H * 2);

  auto psz = [](int Kpad, int Npad) -> size_t {
    return (size_t)(Npad / 16) * (Kpad / 32) * 512 * 2;
  };
  bf16* pE0fWih = (bf16*)alloc(psz(160, 1536));
  bf16* pE0bWih = (bf16*)alloc(psz(160, 1536));
  bf16* pE1fWih = (bf16*)alloc(psz(1024, 1536));
  bf16* pE1bWih = (bf16*)alloc(psz(1024, 1536));
  bf16* pE0fWhh = (bf16*)alloc(psz(512, 1536));
  bf16* pE0bWhh = (bf16*)alloc(psz(512, 1536));
  bf16* pE1fWhh = (bf16*)alloc(psz(512, 1536));
  bf16* pE1bWhh = (bf16*)alloc(psz(512, 1536));
  bf16* pMem    = (bf16*)alloc(psz(1024, 512));
  bf16* pQuery  = (bf16*)alloc(psz(512, 512));
  bf16* pLocD   = (bf16*)alloc(psz(32, 512));
  bf16* pD0Wih  = (bf16*)alloc(psz(672, 1536));
  bf16* pD0Whh  = (bf16*)alloc(psz(512, 1536));
  bf16* pD1Wih  = (bf16*)alloc(psz(512, 1536));
  bf16* pD1Whh  = (bf16*)alloc(psz(512, 1536));
  bf16* pCls1   = (bf16*)alloc(psz(1024, 512));
  bf16* pCls2   = (bf16*)alloc(psz(512, 80));
  bf16* pH2h    = (bf16*)alloc(psz(2048, 1024));

  auto pack = [&](const float* src, bf16* dst, int K, int N, int Kpad, int Npad) {
    int total = (Npad / 16) * (Kpad / 32) * 512;
    k_pack<<<(total + 255) / 256, 256, 0, stream>>>(src, dst, K, N, Kpad, Npad);
  };
  auto gemm = [&](const bf16* A, int lda, const bf16* Wp, const float* bias,
                  float* C, int ldc, bf16* Cb, int ldcb,
                  int rows, int ksteps, int ntiles, int act) {
    int tiles = (rows / 16) * ntiles;
    k_gemm<<<(tiles + 7) / 8, 256, 0, stream>>>(A, lda, Wp, bias, C, ldc, Cb, ldcb,
                                                rows, ksteps, ntiles, act);
  };

  // ---- pack all weights ----
  pack(e0fWih, pE0fWih, 144, 1536, 160, 1536);
  pack(e0bWih, pE0bWih, 144, 1536, 160, 1536);
  pack(e1fWih, pE1fWih, 1024, 1536, 1024, 1536);
  pack(e1bWih, pE1bWih, 1024, 1536, 1024, 1536);
  pack(e0fWhh, pE0fWhh, 512, 1536, 512, 1536);
  pack(e0bWhh, pE0bWhh, 512, 1536, 512, 1536);
  pack(e1fWhh, pE1fWhh, 512, 1536, 512, 1536);
  pack(e1bWhh, pE1bWhh, 512, 1536, 512, 1536);
  pack(memW,   pMem,    1024, 512, 1024, 512);
  pack(queryW, pQuery,  512, 512, 512, 512);
  pack(locDW,  pLocD,   32, 512, 32, 512);
  pack(d0Wih,  pD0Wih,  656, 1536, 672, 1536);
  pack(d0Whh,  pD0Whh,  512, 1536, 512, 1536);
  pack(d1Wih,  pD1Wih,  512, 1536, 512, 1536);
  pack(d1Whh,  pD1Whh,  512, 1536, 512, 1536);
  pack(clsW1,  pCls1,   1024, 512, 1024, 512);
  pack(clsW2,  pCls2,   512, 80, 512, 80);
  pack(h2hW,   pH2h,    2048, 1024, 2048, 1024);

  // ---- encoder ----
  k_prep_xin<<<(ROWS_ENC * 160 + 255) / 256, 256, 0, stream>>>(x, xin);
  // layer 0 input GEMMs (gi includes bih)
  gemm(xin, 160, pE0fWih, e0fbih, giF, G3, nullptr, 0, ROWS_ENC, 5, 96, 0);
  gemm(xin, 160, pE0bWih, e0bbih, giB, G3, nullptr, 0, ROWS_ENC, 5, 96, 0);
  k_gru_rec<<<2, 1024, 262144, stream>>>(giF, giB, pE0fWhh, pE0bWhh, e0fbhh, e0bbhh,
                                         ycur, hflat, 0, 512);
  // layer 1 input GEMMs on concat(yf,yb)
  gemm(ycur, 1024, pE1fWih, e1fbih, giF, G3, nullptr, 0, ROWS_ENC, 32, 96, 0);
  gemm(ycur, 1024, pE1bWih, e1bbih, giB, G3, nullptr, 0, ROWS_ENC, 32, 96, 0);
  k_gru_rec<<<2, 1024, 262144, stream>>>(giF, giB, pE1fWhh, pE1bWhh, e1fbhh, e1bbhh,
                                         ycur, hflat, 1024, 1536);
  // attn_kv = encoder_outputs @ mem_W^T   ((T,B,512))
  gemm(ycur, 1024, pMem, nullptr, kv, H, nullptr, 0, ROWS_ENC, 32, 32, 0);
  // hdec init = flat @ h2h^T + b
  gemm(hflat, 2048, pH2h, h2hb, hdecT, 1024, nullptr, 0, B, 64, 64, 0);
  k_split_hdec<<<(B * 2 * H + 255) / 256, 256, 0, stream>>>(hdecT, hd, hdb);
  k_zero<<<(ROWS_ENC + 255) / 256, 256, 0, stream>>>(aBuf, ROWS_ENC);
  k_zero<<<(ROWS_ENC + 255) / 256, 256, 0, stream>>>(acBuf, ROWS_ENC);

  // ---- decoder ----
  for (int t = 0; t < T_OUT; ++t) {
    // q = hd[1] @ query^T
    gemm(hdb + B * H, H, pQuery, nullptr, qbuf, H, nullptr, 0, B, 16, 32, 0);
    k_loc_conv<<<(ROWS_ENC * NFILT + 255) / 256, 256, 0, stream>>>(aBuf, acBuf, locW, locBuf);
    k_energy<<<(ROWS_ENC / 16) / 8, 256, 0, stream>>>(locBuf, pLocD, qbuf, kv, vW, eBuf,
                                                      ROWS_ENC / 16);
    k_softmax_ctx<<<B, 256, 0, stream>>>(eBuf, aBuf, acBuf, kv, ctx, feat, decin);
    k_build_decin<<<(B * 672 + 255) / 256, 256, 0, stream>>>(decin, out, t);
    // decoder GRU layer 0
    gemm(decin, 672, pD0Wih, d0bih, giD, G3, nullptr, 0, B, 21, 96, 0);
    gemm(hdb, H, pD0Whh, d0bhh, ghD, G3, nullptr, 0, B, 16, 96, 0);
    k_gate<<<(B * H + 255) / 256, 256, 0, stream>>>(giD, ghD, hd, hdb, nullptr, 0);
    // decoder GRU layer 1 (input = updated hd0)
    gemm(hdb, H, pD1Wih, d1bih, giD, G3, nullptr, 0, B, 16, 96, 0);
    gemm(hdb + B * H, H, pD1Whh, d1bhh, ghD, G3, nullptr, 0, B, 16, 96, 0);
    k_gate<<<(B * H + 255) / 256, 256, 0, stream>>>(giD, ghD, hd + B * H, hdb + B * H,
                                                    feat, 1024);
    // classifier: hid = elu(feat @ W1^T + b1); logits -> d_out[:, t, :]
    gemm(feat, 1024, pCls1, clsb1, hid, H, hidb, H, B, 32, 32, 1);
    gemm(hidb, H, pCls2, clsb2, out + (size_t)t * NC, T_OUT * NC, nullptr, 0, B, 16, 5, 0);
  }
}
